// AutoRegressiveLSTM_12189117186530
// MI455X (gfx1250) — matrix operand order repaired and checked by reading, not compiled, not run on hardware
//
#include <hip/hip_runtime.h>

typedef unsigned short u16;
typedef __attribute__((ext_vector_type(16))) __bf16 v16bf;
typedef __attribute__((ext_vector_type(8)))  float  v8f;

#define BSZ   128
#define TSZ   1024
#define ISZ   128
#define HSZ   1024
#define OSZ   128
#define KTOT  1152          // I + H (activation row length)
#define NWG   72            // 64 gate WGs + 8 out WGs
#define NGATEWG 64
#define CHK   64            // K columns staged into LDS per chunk (gate path)
#define NCHK  (KTOT / CHK)  // 18 chunks
#define ASTR  72            // gate LDS row stride (elems): 36 dwords -> bank-tiled
#define OSTR  136           // out LDS row stride (elems): 68 dwords -> bank-tiled

// ---------------- helpers ----------------

__device__ __forceinline__ u16 f2bf(float f) {
  unsigned u = __builtin_bit_cast(unsigned, f);
  unsigned r = u + 0x7FFFu + ((u >> 16) & 1u);   // round-to-nearest-even
  return (u16)(r >> 16);
}

struct U32x8 { uint4 lo, hi; };

__device__ __forceinline__ v16bf load_frag(const u16* p0, const u16* p1) {
  U32x8 r;
  r.lo = *(const uint4*)p0;
  r.hi = *(const uint4*)p1;
  return __builtin_bit_cast(v16bf, r);
}

__device__ __forceinline__ float fsigmoid(float x) { return 1.f / (1.f + __expf(-x)); }
__device__ __forceinline__ float ftanh(float x) {
  float e = __expf(-2.f * x);
  return (1.f - e) / (1.f + e);
}

// packed gate-column order: p = g*256 + u_local*4 + gtype
__device__ __forceinline__ int orig_row(int p) {
  int g = p >> 8, n = p & 255;
  return (n & 3) * HSZ + g * 64 + (n >> 2);
}

// ---------------- precompute kernels ----------------

__global__ void k_pack_wcomb(const float* __restrict__ Wih, const float* __restrict__ Whh,
                             const float* __restrict__ Wout, u16* __restrict__ Wcomb) {
  int idx = blockIdx.x * 256 + threadIdx.x;
  if (idx >= 4096 * KTOT) return;
  int p = idx / KTOT, j = idx % KTOT;
  int r = orig_row(p);
  float v;
  if (j < ISZ) {
    v = Wih[r * 256 + j];
  } else {
    int c = j - ISZ;
    float acc = Whh[r * HSZ + c];
    const float* wrow = Wih + r * 256 + ISZ;
    for (int k = 0; k < OSZ; ++k) acc += wrow[k] * Wout[k * HSZ + c];
    v = acc;
  }
  Wcomb[p * KTOT + j] = f2bf(v);
}

__global__ void k_bias(const float* __restrict__ Wih, const float* __restrict__ bih,
                       const float* __restrict__ bhh, const float* __restrict__ bout,
                       float* __restrict__ bias_eff) {
  int p = blockIdx.x * 256 + threadIdx.x;
  if (p >= 4096) return;
  int r = orig_row(p);
  float acc = bih[r] + bhh[r];
  for (int k = 0; k < OSZ; ++k) acc += Wih[r * 256 + ISZ + k] * bout[k];
  bias_eff[p] = acc;
}

__global__ void k_z0(const float* __restrict__ Wih, const float* __restrict__ out0,
                     const float* __restrict__ bout, float* __restrict__ z0) {
  int idx = blockIdx.x * 256 + threadIdx.x;
  if (idx >= BSZ * 4096) return;
  int b = idx >> 12, p = idx & 4095;
  int r = orig_row(p);
  float acc = 0.f;
  for (int k = 0; k < OSZ; ++k)
    acc += (out0[b * OSZ + k] - bout[k]) * Wih[r * 256 + ISZ + k];
  z0[b * 4096 + p] = acc;
}

__global__ void k_pack_wout(const float* __restrict__ Wout, u16* __restrict__ Woutb) {
  int idx = blockIdx.x * 256 + threadIdx.x;
  if (idx >= OSZ * HSZ) return;
  Woutb[idx] = f2bf(Wout[idx]);
}

__global__ void k_initL(const float* __restrict__ x, const int* __restrict__ seq,
                        const float* __restrict__ hn, u16* __restrict__ L0) {
  int idx = blockIdx.x * 256 + threadIdx.x;
  if (idx >= BSZ * KTOT) return;
  int b = idx / KTOT, j = idx % KTOT;
  float v;
  if (j < ISZ) v = x[(size_t)b * TSZ * ISZ + j] * ((0 < seq[b]) ? 1.f : 0.f);
  else         v = hn[b * HSZ + (j - ISZ)];
  L0[idx] = f2bf(v);
}

// ---------------- persistent LSTM kernel ----------------

__launch_bounds__(256)
__global__ void k_lstm(const u16* __restrict__ Wcomb, const u16* __restrict__ Woutb,
                       const float* __restrict__ bias_eff, const float* __restrict__ z0,
                       u16* __restrict__ L0, u16* __restrict__ L1,
                       const float* __restrict__ x, const int* __restrict__ seq,
                       const float* __restrict__ cn, const float* __restrict__ bout,
                       float* __restrict__ ys, unsigned* __restrict__ bar) {
  __shared__ float lds_c[64 * 128];      // 32KB: gate WGs' persistent cell state
  __shared__ u16   lds_a[BSZ * ASTR];    // 18KB: staged activation chunk (both roles)

  const int wg   = blockIdx.x;
  const int tid  = threadIdx.x;
  const int lane = tid & 31;
  const int w    = tid >> 5;
  const bool isGate = (wg < NGATEWG);
  const int g = wg;

  float biasreg[2] = {0.f, 0.f};
  if (isGate) {
    for (int i = tid; i < 64 * 128; i += 256) {
      int u = i >> 7, b = i & 127;
      lds_c[i] = cn[b * HSZ + g * 64 + u];
    }
    biasreg[0] = bias_eff[g * 256 + w * 32 +  0 + (lane & 15)];
    biasreg[1] = bias_eff[g * 256 + w * 32 + 16 + (lane & 15)];
  }
  __syncthreads();

  for (int t = 0; t <= TSZ; ++t) {
    const u16* Lc = (t & 1) ? L1 : L0;   // [x_t*m_t | h_{t-1}]
    u16*       Ln = (t & 1) ? L0 : L1;

    if (isGate) {
      if (t < TSZ) {
        v8f acc[8][2] = {};
        const u16* wbase = Wcomb + (size_t)(g * 256 + w * 32 + (lane & 15)) * KTOT
                                 + ((lane >> 4) << 4);
        const u16* labase = lds_a + (lane & 15) * ASTR + ((lane >> 4) << 4);

        for (int cc = 0; cc < NCHK; ++cc) {
          __syncthreads();   // previous chunk fully consumed
          // cooperative stage: 128 rows x 64 K (bf16) -> LDS (padded rows)
#pragma unroll
          for (int r = 0; r < 4; ++r) {
            int u0 = tid + r * 256;
            int row = u0 >> 3, seg = (u0 & 7) << 3;
            uint4 d = *(const uint4*)(Lc + (size_t)row * KTOT + cc * CHK + seg);
            *(uint4*)(lds_a + row * ASTR + seg) = d;
          }
          // preload B fragments for both k-steps (global latency hides behind barrier)
          v16bf bfr[2][2];
#pragma unroll
          for (int kk = 0; kk < 2; ++kk) {
            int k0 = cc * CHK + kk * 32;
            bfr[kk][0] = load_frag(wbase + k0,             wbase + k0 + 8);
            bfr[kk][1] = load_frag(wbase + 16 * KTOT + k0, wbase + 16 * KTOT + k0 + 8);
          }
          __syncthreads();   // staged data visible
#pragma unroll
          for (int kk = 0; kk < 2; ++kk) {
            int lk = kk * 32;
#pragma unroll
            for (int half = 0; half < 2; ++half) {
              v16bf afr[4];
#pragma unroll
              for (int m4 = 0; m4 < 4; ++m4) {
                const u16* ap = labase + (half * 4 + m4) * 16 * ASTR + lk;
                afr[m4] = load_frag(ap, ap + 8);   // ds_load_b128 x2
              }
#pragma unroll
              for (int m4 = 0; m4 < 4; ++m4) {
                int mt = half * 4 + m4;
                acc[mt][0] = __builtin_amdgcn_wmma_f32_16x16x32_bf16(
                    false, afr[m4], false, bfr[kk][0], (short)0, acc[mt][0], false, false);
                acc[mt][1] = __builtin_amdgcn_wmma_f32_16x16x32_bf16(
                    false, afr[m4], false, bfr[kk][1], (short)0, acc[mt][1], false, false);
              }
            }
          }
        }

        // cell update: i/f/g/o for one hidden unit live in a 4-lane group
        const int gt = lane & 3;
        for (int nt = 0; nt < 2; ++nt) {
          int colw  = w * 32 + nt * 16 + (lane & 15);
          int u_loc = colw >> 2;
          int pg    = g * 256 + colw;
          float bia = biasreg[nt];
#pragma unroll
          for (int mt = 0; mt < 8; ++mt) {
            v8f v = acc[mt][nt];
#pragma unroll
            for (int j = 0; j < 8; ++j) {
              int b = mt * 16 + j + ((lane >> 4) << 3);
              float pre = v[j] + bia;
              if (t == 0) pre += z0[b * 4096 + pg];
              float act = (gt == 2) ? ftanh(pre) : fsigmoid(pre);
              float v1 = __shfl_xor(act, 1, 32);
              float v2 = __shfl_xor(act, 2, 32);
              float v3 = __shfl_xor(act, 3, 32);
              // valid on gt==0 lanes: i=act, f=v1, g=v2, o=v3
              float cold = lds_c[u_loc * 128 + b];
              float cnew = v1 * cold + act * v2;
              float hnew = v3 * ftanh(cnew);
              if (gt == 0) {
                lds_c[u_loc * 128 + b] = cnew;
                Ln[b * KTOT + ISZ + g * 64 + u_loc] = f2bf(hnew);
              }
            }
          }
        }

        // stage x_{t+1}*m_{t+1}: each gate-WG owns 2 input columns
        if (t + 1 < TSZ) {
          int b = tid >> 1, col = (g << 1) + (tid & 1);
          float m = ((t + 1) < seq[b]) ? 1.f : 0.f;
          float xv = x[(size_t)b * TSZ * ISZ + (size_t)(t + 1) * ISZ + col];
          Ln[b * KTOT + col] = f2bf(xv * m);
        }
      }
    } else if (t >= 1) {
      // out WG q: batches [16q,16q+16), wave w: out-cols [16w,16w+16), full K.
      // All waves share one 16-row A slab -> LDS staged per 128-K chunk.
      int q = wg - NGATEWG;
      v8f oacc = {};
      const u16* wob = Woutb + (size_t)(w * 16 + (lane & 15)) * HSZ + ((lane >> 4) << 4);
      const u16* labase = lds_a + (lane & 15) * OSTR + ((lane >> 4) << 4);
      for (int cc = 0; cc < 8; ++cc) {
        __syncthreads();   // previous chunk consumed
        {
          int row = tid >> 4, seg = (tid & 15) << 3;   // 16 rows x 16 b128-segs
          uint4 d = *(const uint4*)(Lc + (size_t)(q * 16 + row) * KTOT + ISZ + cc * 128 + seg);
          *(uint4*)(lds_a + row * OSTR + seg) = d;
        }
        v16bf bfr[4];
#pragma unroll
        for (int kk = 0; kk < 4; ++kk) {
          int k0 = cc * 128 + kk * 32;
          bfr[kk] = load_frag(wob + k0, wob + k0 + 8);
        }
        __syncthreads();   // staged data visible
#pragma unroll
        for (int kk = 0; kk < 4; ++kk) {
          const u16* ap = labase + kk * 32;
          v16bf a = load_frag(ap, ap + 8);   // ds_load_b128 x2
          oacc = __builtin_amdgcn_wmma_f32_16x16x32_bf16(
              false, a, false, bfr[kk], (short)0, oacc, false, false);
        }
      }
      // direct writeback: C tile is the final 16x16 out block (bias + mask)
      int tm1 = t - 1;
      int bl  = (lane >> 4) << 3;
      int col = w * 16 + (lane & 15);
      float bo = bout[col];
#pragma unroll
      for (int j = 0; j < 8; ++j) {
        int b = q * 16 + j + bl;
        float m = (tm1 < seq[b]) ? 1.f : 0.f;
        ys[(size_t)b * TSZ * OSZ + (size_t)tm1 * OSZ + col] = (oacc[j] + bo) * m;
      }
    }

    // device-wide barrier (monotonic counter, no reset race)
    if (t < TSZ) {
      __syncthreads();
      if (tid == 0) {
        __threadfence();
        atomicAdd(bar, 1u);
        unsigned target = (unsigned)NWG * (unsigned)(t + 1);
        while (__hip_atomic_load(bar, __ATOMIC_ACQUIRE, __HIP_MEMORY_SCOPE_AGENT) < target)
          __builtin_amdgcn_s_sleep(1);
      }
      __syncthreads();
      __threadfence();
    }
  }
}

// ---------------- launch ----------------

extern "C" void kernel_launch(void* const* d_in, const int* in_sizes, int n_in,
                              void* d_out, int out_size, void* d_ws, size_t ws_size,
                              hipStream_t stream) {
  (void)in_sizes; (void)n_in; (void)out_size; (void)ws_size;
  const float* x    = (const float*)d_in[0];
  const int*   seq  = (const int*)d_in[1];
  const float* hn   = (const float*)d_in[2];
  const float* cn   = (const float*)d_in[3];
  const float* out0 = (const float*)d_in[4];
  const float* Wih  = (const float*)d_in[5];
  const float* Whh  = (const float*)d_in[6];
  const float* bih  = (const float*)d_in[7];
  const float* bhh  = (const float*)d_in[8];
  const float* Wout = (const float*)d_in[9];
  const float* bout = (const float*)d_in[10];
  float* ys = (float*)d_out;

  char* ws = (char*)d_ws;
  size_t off = 0;
  unsigned* bar   = (unsigned*)(ws + off); off += 256;
  u16* Wcomb      = (u16*)(ws + off);      off += (size_t)4096 * KTOT * 2;
  u16* Woutb      = (u16*)(ws + off);      off += (size_t)OSZ * HSZ * 2;
  float* bias_eff = (float*)(ws + off);    off += 4096 * 4;
  float* z0       = (float*)(ws + off);    off += (size_t)BSZ * 4096 * 4;
  u16* L0         = (u16*)(ws + off);      off += (size_t)BSZ * KTOT * 2;
  u16* L1         = (u16*)(ws + off);      off += (size_t)BSZ * KTOT * 2;

  hipMemsetAsync(bar, 0, 256, stream);
  k_pack_wcomb<<<(4096 * KTOT + 255) / 256, 256, 0, stream>>>(Wih, Whh, Wout, Wcomb);
  k_bias<<<16, 256, 0, stream>>>(Wih, bih, bhh, bout, bias_eff);
  k_z0<<<(BSZ * 4096 + 255) / 256, 256, 0, stream>>>(Wih, out0, bout, z0);
  k_pack_wout<<<(OSZ * HSZ + 255) / 256, 256, 0, stream>>>(Wout, Woutb);
  k_initL<<<(BSZ * KTOT + 255) / 256, 256, 0, stream>>>(x, seq, hn, L0);
  k_lstm<<<NWG, 256, 0, stream>>>(Wcomb, Woutb, bias_eff, z0, L0, L1,
                                  x, seq, cn, bout, ys, bar);
}
